// MoEMultiBranchFFN_26671746908373
// MI455X (gfx1250) — compile-verified, matrix operand order checked by reading, not run on hardware
//
#include <hip/hip_runtime.h>
#include <hip/hip_bf16.h>
#include <math.h>

typedef __attribute__((ext_vector_type(16))) __bf16 v16bf;
typedef __attribute__((ext_vector_type(8)))  float  v8f;

#define BB  2
#define TT  2048
#define DD  512
#define HH  2048
#define EE  8
#define BRR 2
#define NN  (BB*TT)   /* 4096 tokens */

// ---------------------------------------------------------------------------
// Tiled transpose + f32->bf16 convert:  in [batch][R][C] f32 -> out [batch][C][R] bf16
// grid = (C/32, R/32, batch), block = 256
// ---------------------------------------------------------------------------
__global__ __launch_bounds__(256) void tpose_k(const float* __restrict__ in,
                                               __bf16* __restrict__ out,
                                               int R, int C) {
    __shared__ float tile[32][33];
    const int b  = blockIdx.z;
    const int c0 = blockIdx.x * 32, r0 = blockIdx.y * 32;
    const float* ip = in  + (size_t)b * R * C;
    __bf16*      op = out + (size_t)b * R * C;
    const int tx = threadIdx.x & 31, ty = threadIdx.x >> 5;
    #pragma unroll
    for (int i = 0; i < 32; i += 8)
        tile[ty + i][tx] = ip[(size_t)(r0 + ty + i) * C + (c0 + tx)];
    __syncthreads();
    #pragma unroll
    for (int i = 0; i < 32; i += 8)
        op[(size_t)(c0 + ty + i) * R + (r0 + tx)] = (__bf16)tile[tx][ty + i];
}

// ---------------------------------------------------------------------------
// LayerNorm (both branches) + gate logits + softmax + top-2.
// wave-per-token; block = 256 (8 tokens), grid = NN/8.
// ---------------------------------------------------------------------------
__global__ __launch_bounds__(256) void ln_gate_k(
    const float* __restrict__ x, const float* __restrict__ ln_g,
    const float* __restrict__ ln_b, const float* __restrict__ gate_w,
    const float* __restrict__ gate_b, __bf16* __restrict__ hfb,
    float* __restrict__ gatev) {
    const int wid = threadIdx.x >> 5, lane = threadIdx.x & 31;
    const int n = blockIdx.x * 8 + wid;
    const float* xr = x + (size_t)n * DD;
    float v[16];
    float s = 0.f;
    #pragma unroll
    for (int i = 0; i < 16; ++i) { v[i] = xr[lane + i * 32]; s += v[i]; }
    #pragma unroll
    for (int o = 16; o > 0; o >>= 1) s += __shfl_xor(s, o, 32);
    const float mu = s * (1.f / DD);
    float q = 0.f;
    #pragma unroll
    for (int i = 0; i < 16; ++i) { float d = v[i] - mu; q += d * d; }
    #pragma unroll
    for (int o = 16; o > 0; o >>= 1) q += __shfl_xor(q, o, 32);
    const float rs = rsqrtf(q * (1.f / DD) + 1e-5f);

    for (int r = 0; r < BRR; ++r) {
        float lg[8] = {0.f, 0.f, 0.f, 0.f, 0.f, 0.f, 0.f, 0.f};
        #pragma unroll
        for (int i = 0; i < 16; ++i) {
            const int d = lane + i * 32;
            const float hv = (v[i] - mu) * rs * ln_g[r * DD + d] + ln_b[r * DD + d];
            hfb[((size_t)r * NN + n) * DD + d] = (__bf16)hv;
            const float* gw = gate_w + ((size_t)r * DD + d) * EE;
            #pragma unroll
            for (int e = 0; e < 8; ++e) lg[e] += hv * gw[e];
        }
        #pragma unroll
        for (int e = 0; e < 8; ++e)
            #pragma unroll
            for (int o = 16; o > 0; o >>= 1) lg[e] += __shfl_xor(lg[e], o, 32);
        if (lane == 0) {
            float mx = -1e30f;
            #pragma unroll
            for (int e = 0; e < 8; ++e) { lg[e] += gate_b[r * EE + e]; mx = fmaxf(mx, lg[e]); }
            float p[8], den = 0.f;
            #pragma unroll
            for (int e = 0; e < 8; ++e) { p[e] = expf(lg[e] - mx); den += p[e]; }
            const float inv = 1.f / den;
            int i0 = 0;
            #pragma unroll
            for (int e = 1; e < 8; ++e) if (p[e] > p[i0]) i0 = e;
            int i1 = (i0 == 0) ? 1 : 0;
            #pragma unroll
            for (int e = 0; e < 8; ++e) if (e != i0 && p[e] > p[i1]) i1 = e;
            float* g = gatev + ((size_t)r * NN + n) * EE;
            #pragma unroll
            for (int e = 0; e < 8; ++e)
                g[e] = (e == i0) ? p[i0] * inv : ((e == i1) ? p[i1] * inv : 0.f);
        }
    }
}

// ---------------------------------------------------------------------------
// Embedding MLP: styl[b][j] = (silu(emb[b]) @ emb_w)[j] + emb_b[j]
// ---------------------------------------------------------------------------
__global__ __launch_bounds__(256) void emb_k(const float* __restrict__ emb,
                                             const float* __restrict__ emb_w,
                                             const float* __restrict__ emb_b,
                                             float* __restrict__ styl) {
    const int j = blockIdx.x * 256 + threadIdx.x;      // 0..2047
    const int b = j >> 10, jj = j & 1023;
    float a = emb_b[jj];
    for (int t = 0; t < 512; ++t) {
        const float e  = emb[b * 512 + t];
        const float se = e / (1.f + expf(-e));
        a += se * emb_w[(size_t)t * 1024 + jj];
    }
    styl[j] = a;
}

// ---------------------------------------------------------------------------
// Post kernel: o = acc/BR; LN(pn); *(1+scale)+shift; SiLU; bf16 store.
// ---------------------------------------------------------------------------
__global__ __launch_bounds__(256) void post_k(const float* __restrict__ acc,
                                              const float* __restrict__ pn_g,
                                              const float* __restrict__ pn_b,
                                              const float* __restrict__ styl,
                                              __bf16* __restrict__ h2b) {
    const int wid = threadIdx.x >> 5, lane = threadIdx.x & 31;
    const int n = blockIdx.x * 8 + wid;
    const int b = n >> 11;                     // T = 2048
    const float* ar = acc + (size_t)n * DD;
    float v[16];
    float s = 0.f;
    #pragma unroll
    for (int i = 0; i < 16; ++i) { v[i] = ar[lane + i * 32] * (1.f / BRR); s += v[i]; }
    #pragma unroll
    for (int o = 16; o > 0; o >>= 1) s += __shfl_xor(s, o, 32);
    const float mu = s * (1.f / DD);
    float q = 0.f;
    #pragma unroll
    for (int i = 0; i < 16; ++i) { float d = v[i] - mu; q += d * d; }
    #pragma unroll
    for (int o = 16; o > 0; o >>= 1) q += __shfl_xor(q, o, 32);
    const float rs = rsqrtf(q * (1.f / DD) + 1e-5f);
    #pragma unroll
    for (int i = 0; i < 16; ++i) {
        const int d = lane + i * 32;
        float y = (v[i] - mu) * rs * pn_g[d] + pn_b[d];
        const float sc = styl[b * 1024 + d];
        const float sh = styl[b * 1024 + 512 + d];
        y = y * (1.f + sc) + sh;
        y = y / (1.f + expf(-y));              // SiLU
        h2b[(size_t)n * DD + d] = (__bf16)y;
    }
}

// ---------------------------------------------------------------------------
// WMMA bf16 GEMM with async-to-LDS double buffering.
//   C(N x NC) = A(N x K) @ B(NC x K)^T   (B stored col-as-row, K-contiguous)
// block tile 128x128, K-step 32; 8 waves, each wave -> 64x32 (eight 16x16).
// Tiles are DMA'd memory->LDS with GLOBAL_LOAD_ASYNC_TO_LDS_B128 (ASYNCcnt),
// buffer b^1 fills while WMMAs consume buffer b.
// EPI 0: +b1, exact GELU -> bf16      (hidden buffer)
// EPI 1: +b2, *gate[e], beta-acc f32  (expert combine)
// EPI 2: +out_b, +x residual -> f32   (final output)
// ---------------------------------------------------------------------------
__device__ inline v16bf ldfrag(const __bf16* s, int row, int kb) {
    // per-lane K layout of V_WMMA_*_16X16X32: elems 0..7 = K kb..kb+7,
    // elems 8..15 = K kb+16..kb+23 (kb = 0 for lanes 0-15, 8 for 16-31)
    v16bf f;
    const __bf16* p = s + row * 32 + kb;
    ((float4*)&f)[0] = *(const float4*)p;
    ((float4*)&f)[1] = *(const float4*)(p + 16);
    return f;
}

__device__ inline void async_copy32(unsigned int lds, unsigned long long gaddr) {
    // per-lane 32B memory->LDS; INST_OFFSET applies to both LDS and global addr
    asm volatile("global_load_async_to_lds_b128 %0, %1, off"
                 :: "v"(lds), "v"(gaddr) : "memory");
    asm volatile("global_load_async_to_lds_b128 %0, %1, off offset:16"
                 :: "v"(lds), "v"(gaddr) : "memory");
}

template <int EPI>
__global__ __launch_bounds__(256) void gemm_k(
    const __bf16* __restrict__ A, const __bf16* __restrict__ Bm,
    int K, int NC, const float* __restrict__ bias,
    const float* __restrict__ gatev, int eidx, int beta,
    float* __restrict__ Cf, __bf16* __restrict__ Cbf,
    const float* __restrict__ Xres) {
    __shared__ __align__(16) __bf16 As[2][128 * 32];
    __shared__ __align__(16) __bf16 Bs[2][128 * 32];
    const int tid = threadIdx.x, wid = tid >> 5, lane = tid & 31;
    const int wm = wid & 1, wn = wid >> 1;
    const int m0 = blockIdx.x * 128, n0 = blockIdx.y * 128;
    const int lr = lane & 15, kb = (lane < 16) ? 0 : 8;
    v8f acc[4][2];
    #pragma unroll
    for (int i = 0; i < 4; ++i)
        #pragma unroll
        for (int j = 0; j < 2; ++j) acc[i][j] = (v8f)(0.0f);

    // cooperative tile mapping: thread owns 32B half-row (16 bf16)
    const int lrow = tid >> 1, lko = (tid & 1) * 16;
    const __bf16* gA = A  + (size_t)(m0 + lrow) * K + lko;
    const __bf16* gB = Bm + (size_t)(n0 + lrow) * K + lko;
    // low 32 bits of a generic __shared__ pointer == LDS byte offset
    const unsigned int lA0 = (unsigned int)(uintptr_t)(&As[0][lrow * 32 + lko]);
    const unsigned int lB0 = (unsigned int)(uintptr_t)(&Bs[0][lrow * 32 + lko]);

    // prologue: fill buffer 0
    async_copy32(lA0, (unsigned long long)(uintptr_t)gA);
    async_copy32(lB0, (unsigned long long)(uintptr_t)gB);
    asm volatile("s_wait_asynccnt 0x0" ::: "memory");
    __syncthreads();

    int buf = 0;
    for (int k0 = 0; k0 < K; k0 += 32) {
        if (k0 + 32 < K) {  // kick off DMA for the next K-slice into buf^1
            const unsigned int boff = (unsigned int)((buf ^ 1) * 128 * 32 * 2);
            async_copy32(lA0 + boff, (unsigned long long)(uintptr_t)(gA + k0 + 32));
            async_copy32(lB0 + boff, (unsigned long long)(uintptr_t)(gB + k0 + 32));
        }
        const __bf16* as = As[buf];
        const __bf16* bs = Bs[buf];
        v16bf bf[2];
        #pragma unroll
        for (int j = 0; j < 2; ++j) bf[j] = ldfrag(bs, wn * 32 + j * 16 + lr, kb);
        #pragma unroll
        for (int i = 0; i < 4; ++i) {
            v16bf af = ldfrag(as, wm * 64 + i * 16 + lr, kb);
            #pragma unroll
            for (int j = 0; j < 2; ++j)
                acc[i][j] = __builtin_amdgcn_wmma_f32_16x16x32_bf16(
                    false, af, false, bf[j], (short)0, acc[i][j], false, false);
        }
        asm volatile("s_wait_asynccnt 0x0" ::: "memory");
        __syncthreads();
        buf ^= 1;
    }

    const int rowadd = (lane < 16) ? 0 : 8;   // C layout: M = g (+8 hi half), N = lane%16
    #pragma unroll
    for (int i = 0; i < 4; ++i)
        #pragma unroll
        for (int j = 0; j < 2; ++j) {
            const int col = n0 + wn * 32 + j * 16 + lr;
            const float bcol = bias[col];
            #pragma unroll
            for (int g = 0; g < 8; ++g) {
                const int row = m0 + wm * 64 + i * 16 + g + rowadd;
                const float c = acc[i][j][g] + bcol;
                const size_t idx = (size_t)row * NC + col;
                if constexpr (EPI == 0) {
                    const float gl = 0.5f * c * (1.f + erff(c * 0.70710678118654752f));
                    Cbf[idx] = (__bf16)gl;
                } else if constexpr (EPI == 1) {
                    float vo = c * gatev[(size_t)row * EE + eidx];
                    if (beta) vo += Cf[idx];
                    Cf[idx] = vo;
                } else {
                    Cf[idx] = c + Xres[idx];
                }
            }
        }
}

// ---------------------------------------------------------------------------
extern "C" void kernel_launch(void* const* d_in, const int* in_sizes, int n_in,
                              void* d_out, int out_size, void* d_ws, size_t ws_size,
                              hipStream_t stream) {
    (void)in_sizes; (void)n_in; (void)out_size; (void)ws_size;
    const float* x      = (const float*)d_in[0];
    const float* emb    = (const float*)d_in[1];
    const float* ln_g   = (const float*)d_in[2];
    const float* ln_b   = (const float*)d_in[3];
    const float* gate_w = (const float*)d_in[4];
    const float* gate_b = (const float*)d_in[5];
    const float* w1     = (const float*)d_in[6];
    const float* b1     = (const float*)d_in[7];
    const float* w2     = (const float*)d_in[8];
    const float* b2     = (const float*)d_in[9];
    const float* emb_w  = (const float*)d_in[10];
    const float* emb_b  = (const float*)d_in[11];
    const float* pn_g   = (const float*)d_in[12];
    const float* pn_b   = (const float*)d_in[13];
    const float* out_w  = (const float*)d_in[14];
    const float* out_b  = (const float*)d_in[15];

    char* wsp = (char*)d_ws; size_t off = 0;
    auto alloc = [&](size_t bytes) -> void* {
        void* p = wsp + off; off = (off + bytes + 255) & ~(size_t)255; return p;
    };
    __bf16* w1t  = (__bf16*)alloc((size_t)BRR * EE * HH * DD * 2);  // [p][H][D]
    __bf16* w2t  = (__bf16*)alloc((size_t)BRR * EE * DD * HH * 2);  // [p][D][H]
    __bf16* owt  = (__bf16*)alloc((size_t)DD * DD * 2);             // [D][D]
    __bf16* hfb  = (__bf16*)alloc((size_t)BRR * NN * DD * 2);       // LN'd acts
    float*  gtv  = (float*) alloc((size_t)BRR * NN * EE * 4);       // gate values
    __bf16* hbuf = (__bf16*)alloc((size_t)NN * HH * 2);             // per-pair hidden
    float*  accb = (float*) alloc((size_t)NN * DD * 4);             // expert accum
    float*  styl = (float*) alloc((size_t)BB * 2 * DD * 4);         // scale|shift
    __bf16* h2b  = (__bf16*)alloc((size_t)NN * DD * 2);             // styled acts

    const dim3 blk(256);
    // one-time weight convert/transpose (f32 -> bf16, K-contiguous per column)
    tpose_k<<<dim3(HH / 32, DD / 32, BRR * EE), blk, 0, stream>>>(w1, w1t, DD, HH);
    tpose_k<<<dim3(DD / 32, HH / 32, BRR * EE), blk, 0, stream>>>(w2, w2t, HH, DD);
    tpose_k<<<dim3(DD / 32, DD / 32, 1),        blk, 0, stream>>>(out_w, owt, DD, DD);

    ln_gate_k<<<dim3(NN / 8), blk, 0, stream>>>(x, ln_g, ln_b, gate_w, gate_b, hfb, gtv);
    emb_k<<<dim3(8), blk, 0, stream>>>(emb, emb_w, emb_b, styl);

    for (int p = 0; p < BRR * EE; ++p) {
        const int r = p >> 3, e = p & 7;
        gemm_k<0><<<dim3(NN / 128, HH / 128), blk, 0, stream>>>(
            hfb + (size_t)r * NN * DD, w1t + (size_t)p * HH * DD, DD, HH,
            b1 + (size_t)p * HH, nullptr, 0, 0, nullptr, hbuf, nullptr);
        gemm_k<1><<<dim3(NN / 128, DD / 128), blk, 0, stream>>>(
            hbuf, w2t + (size_t)p * DD * HH, HH, DD,
            b2 + (size_t)p * DD, gtv + (size_t)r * NN * EE, e, (p > 0) ? 1 : 0,
            accb, nullptr, nullptr);
    }

    post_k<<<dim3(NN / 8), blk, 0, stream>>>(accb, pn_g, pn_b, styl, h2b);
    gemm_k<2><<<dim3(NN / 128, DD / 128), blk, 0, stream>>>(
        h2b, owt, DD, DD, out_b, nullptr, 0, 0, (float*)d_out, nullptr, x);
}
